// Discriminator_30279519436849
// MI455X (gfx1250) — compile-verified, matrix operand order checked by reading, not run on hardware
//
#include <hip/hip_runtime.h>
#include <stdint.h>

#define BATCH 8
#define N0    4096
#define DIM   64
#define KNB   8          // K neighbors
#define NEG   0.01f
#define NT    32         // points per resblock tile
#define LDP   (DIM + 2)  // padded LDS row (halves) to avoid bank conflicts

typedef __attribute__((ext_vector_type(16))) _Float16 v16h;
typedef __attribute__((ext_vector_type(8)))  float    v8f;

__device__ __forceinline__ float leaky(float x) { return x > 0.f ? x : NEG * x; }

// ---------------------------------------------------------------------------
// points = w0 @ xyz  : [64,3] x [b,3,n] -> [b,64,n]
// ---------------------------------------------------------------------------
__global__ void lift_kernel(const float* __restrict__ xyz, const float* __restrict__ w0,
                            float* __restrict__ out, int n) {
  int i = blockIdx.x * blockDim.x + threadIdx.x;
  int b = blockIdx.y;
  if (i >= n) return;
  float x = xyz[((size_t)b * 3 + 0) * n + i];
  float y = xyz[((size_t)b * 3 + 1) * n + i];
  float z = xyz[((size_t)b * 3 + 2) * n + i];
#pragma unroll 4
  for (int c = 0; c < DIM; ++c) {
    out[((size_t)b * DIM + c) * n + i] = w0[c * 3 + 0] * x + w0[c * 3 + 1] * y + w0[c * 3 + 2] * z;
  }
}

// ---------------------------------------------------------------------------
// kNN: for each query (m of them) find 8 nearest refs (n of them).
// d = |q|^2 + |r|^2 - 2 q.r   (reference formula). Refs staged through LDS.
// ---------------------------------------------------------------------------
#define KNN_CH 1024
__global__ void knn_kernel(const float* __restrict__ q, const float* __restrict__ ref,
                           int* __restrict__ idx, int m, int n) {
  __shared__ float rx[KNN_CH], ry[KNN_CH], rz[KNN_CH];
  int b  = blockIdx.y;
  int qi = blockIdx.x * blockDim.x + threadIdx.x;
  bool valid = qi < m;
  float qx = 0.f, qy = 0.f, qz = 0.f, qq = 0.f;
  if (valid) {
    qx = q[((size_t)b * 3 + 0) * m + qi];
    qy = q[((size_t)b * 3 + 1) * m + qi];
    qz = q[((size_t)b * 3 + 2) * m + qi];
    qq = qx * qx + qy * qy + qz * qz;
  }
  float bd[KNB];
  int   bi[KNB];
#pragma unroll
  for (int t = 0; t < KNB; ++t) { bd[t] = 3.4e38f; bi[t] = 0; }

  for (int c0 = 0; c0 < n; c0 += KNN_CH) {
    int cn = (n - c0 < KNN_CH) ? (n - c0) : KNN_CH;
    for (int t = threadIdx.x; t < cn; t += blockDim.x) {
      rx[t] = ref[((size_t)b * 3 + 0) * n + c0 + t];
      ry[t] = ref[((size_t)b * 3 + 1) * n + c0 + t];
      rz[t] = ref[((size_t)b * 3 + 2) * n + c0 + t];
    }
    __syncthreads();
    if (valid) {
      for (int t = 0; t < cn; ++t) {
        float x = rx[t], y = ry[t], z = rz[t];
        float d = qq + (x * x + y * y + z * z) - 2.f * (qx * x + qy * y + qz * z);
        if (d < bd[KNB - 1]) {
          int pos = KNB - 1;
          while (pos > 0 && bd[pos - 1] > d) {
            bd[pos] = bd[pos - 1]; bi[pos] = bi[pos - 1]; --pos;
          }
          bd[pos] = d; bi[pos] = c0 + t;
        }
      }
    }
    __syncthreads();
  }
  if (valid) {
#pragma unroll
    for (int t = 0; t < KNB; ++t) idx[((size_t)b * m + qi) * KNB + t] = bi[t];
  }
}

// ---------------------------------------------------------------------------
// Farthest point sampling: one block per batch. dist[] lives in LDS.
// argmax prefers the lowest index on ties (matches jnp.argmax).
// ---------------------------------------------------------------------------
__global__ void __launch_bounds__(256) fps_kernel(const float* __restrict__ xyz,
                                                  int* __restrict__ sidx, int n, int npoint) {
  __shared__ float dist[N0];
  __shared__ float rv[256];
  __shared__ int   ri[256];
  __shared__ int   s_last;
  int b = blockIdx.x;
  int tid = threadIdx.x;
  for (int i = tid; i < n; i += 256) dist[i] = 1e10f;
  if (tid == 0) { s_last = 0; sidx[(size_t)b * npoint + 0] = 0; }
  __syncthreads();
  for (int it = 1; it < npoint; ++it) {
    int last = s_last;
    float px = xyz[((size_t)b * 3 + 0) * n + last];
    float py = xyz[((size_t)b * 3 + 1) * n + last];
    float pz = xyz[((size_t)b * 3 + 2) * n + last];
    float bestv = -1.f; int besti = n;
    for (int i = tid; i < n; i += 256) {
      float dx = xyz[((size_t)b * 3 + 0) * n + i] - px;
      float dy = xyz[((size_t)b * 3 + 1) * n + i] - py;
      float dz = xyz[((size_t)b * 3 + 2) * n + i] - pz;
      float d  = dx * dx + dy * dy + dz * dz;
      float dm = fminf(dist[i], d);
      dist[i]  = dm;
      if (dm > bestv) { bestv = dm; besti = i; }
    }
    rv[tid] = bestv; ri[tid] = besti;
    __syncthreads();
    for (int s = 128; s > 0; s >>= 1) {
      if (tid < s) {
        float ov = rv[tid + s]; int oi = ri[tid + s];
        if (ov > rv[tid] || (ov == rv[tid] && oi < ri[tid])) { rv[tid] = ov; ri[tid] = oi; }
      }
      __syncthreads();
    }
    if (tid == 0) { s_last = ri[0]; sidx[(size_t)b * npoint + it] = ri[0]; }
    __syncthreads();
  }
}

// ---------------------------------------------------------------------------
__global__ void gather_xyz_kernel(const float* __restrict__ xyz, const int* __restrict__ sidx,
                                  float* __restrict__ out, int n, int npoint) {
  int j = blockIdx.x * blockDim.x + threadIdx.x;
  int b = blockIdx.y;
  if (j >= npoint) return;
  int s = sidx[(size_t)b * npoint + j];
#pragma unroll
  for (int d = 0; d < 3; ++d)
    out[((size_t)b * 3 + d) * npoint + j] = xyz[((size_t)b * 3 + d) * n + s];
}

// new_points[b,c,j] = max_k points[b,c,idx[b,j,k]]
__global__ void pool_max_kernel(const float* __restrict__ pin, const int* __restrict__ idx,
                                float* __restrict__ pout, int n, int npoint) {
  int j = blockIdx.x * blockDim.x + threadIdx.x;
  int b = blockIdx.y;
  if (j >= npoint) return;
  const int* ip = idx + ((size_t)b * npoint + j) * KNB;
  int g[KNB];
#pragma unroll
  for (int t = 0; t < KNB; ++t) g[t] = ip[t];
#pragma unroll 4
  for (int c = 0; c < DIM; ++c) {
    const float* p = pin + ((size_t)b * DIM + c) * n;
    float m = p[g[0]];
#pragma unroll
    for (int t = 1; t < KNB; ++t) m = fmaxf(m, p[g[t]]);
    pout[((size_t)b * DIM + c) * npoint + j] = m;
  }
}

// ---------------------------------------------------------------------------
// WMMA fragment loader.  A (16x32, MxK) and B (32x16, KxN) 16-bit layouts per
// ISA 7.12.2: lanes 0-15 hold M(/N)=lane, VGPR r holds K pair:
//   r<4 : K = 2r      (+8 if lane>=16)
//   r>=4: K = 16+2(r-4) (+8 if lane>=16)
// `row` points at a 64-entry K-contiguous LDS row; pairs read as 32-bit words.
// ---------------------------------------------------------------------------
__device__ __forceinline__ v16h load_frag(const _Float16* row, int lh, int koff) {
  union { v16h v; uint32_t u[8]; } f;
#pragma unroll
  for (int r = 0; r < 8; ++r) {
    int k = ((r < 4) ? (2 * r) : (16 + 2 * (r - 4))) + 8 * lh + koff;
    f.u[r] = *(const uint32_t*)(row + k);
  }
  return f.v;
}

// ---------------------------------------------------------------------------
// One ResGraphConv block:
//   h = leaky(pin); s[:,i] = sum_k h[:, idx[i,k]]
//   pout = (wc@h + wg@s)/9 + pin
// 256 threads = 8 waves; block = (batch, 32-point tile); each wave computes a
// 16x16 output tile via 4x v_wmma_f32_16x16x32_f16 (f32 accumulate).
// FOUR fully independent accumulators (branch x K-half): no WMMA reads any
// other WMMA's D register, so the hazard tables require zero NOP padding and
// the matrix pipe can stream the four ops back-to-back.  The partial tiles
// are reduced with VALU adds in the epilogue.
// ---------------------------------------------------------------------------
__global__ void __launch_bounds__(256) resblock_kernel(
    const float* __restrict__ pin, const int* __restrict__ idx,
    const float* __restrict__ wc, const float* __restrict__ wg,
    float* __restrict__ pout, int n) {
  __shared__ _Float16 sW[2][DIM][LDP];  // f16 weights, [M][K]
  __shared__ _Float16 sH[NT][LDP];      // f16 activations, [point][K=channel]
  __shared__ _Float16 sS[NT][LDP];      // f16 neighbor sums

  int b    = blockIdx.y;
  int base = blockIdx.x * NT;
  int tid  = threadIdx.x;

  // stage weights as f16 (K-contiguous rows)
  for (int e = tid; e < DIM * DIM; e += 256) {
    int r = e >> 6, c = e & 63;
    sW[0][r][c] = (_Float16)wc[e];
    sW[1][r][c] = (_Float16)wg[e];
  }
  // stage H = leaky(pin) tile and S = neighbor sums of H
  for (int e = tid; e < NT * DIM; e += 256) {
    int p = e & (NT - 1);
    int c = e >> 5;
    int gi = base + p;
    const float* prow = pin + ((size_t)b * DIM + c) * n;
    sH[p][c] = (_Float16)leaky(prow[gi]);
    const int* ip = idx + ((size_t)b * n + gi) * KNB;
    float s = 0.f;
#pragma unroll
    for (int t = 0; t < KNB; ++t) s += leaky(prow[ip[t]]);
    sS[p][c] = (_Float16)s;
  }
  __syncthreads();

  int wave = tid >> 5, lane = tid & 31;
  int mtile = wave & 3, ntile = wave >> 2;  // 4 M-tiles x 2 N-tiles
  int l15 = lane & 15, lh = lane >> 4;
  int M    = mtile * 16 + l15;   // weight row this lane supplies (A)
  int pcol = ntile * 16 + l15;   // point this lane supplies (B)

  const _Float16* wcrow = &sW[0][M][0];
  const _Float16* wgrow = &sW[1][M][0];
  const _Float16* hrow  = &sH[pcol][0];
  const _Float16* srow  = &sS[pcol][0];

  // hoist all 8 fragments, then issue 4 fully independent WMMAs
  v16h ac0 = load_frag(wcrow, lh, 0);
  v16h bh0 = load_frag(hrow,  lh, 0);
  v16h ag0 = load_frag(wgrow, lh, 0);
  v16h bs0 = load_frag(srow,  lh, 0);
  v16h ac1 = load_frag(wcrow, lh, 32);
  v16h bh1 = load_frag(hrow,  lh, 32);
  v16h ag1 = load_frag(wgrow, lh, 32);
  v16h bs1 = load_frag(srow,  lh, 32);

  v8f z = {};
  v8f t0 = __builtin_amdgcn_wmma_f32_16x16x32_f16(false, ac0, false, bh0, (short)0, z, false, false);
  v8f t1 = __builtin_amdgcn_wmma_f32_16x16x32_f16(false, ag0, false, bs0, (short)0, z, false, false);
  v8f t2 = __builtin_amdgcn_wmma_f32_16x16x32_f16(false, ac1, false, bh1, (short)0, z, false, false);
  v8f t3 = __builtin_amdgcn_wmma_f32_16x16x32_f16(false, ag1, false, bs1, (short)0, z, false, false);

  v8f acc = (t0 + t1) + (t2 + t3);

  // C/D layout: VGPR v -> M = 16*mtile + v + 8*lh, N = l15
  const float inv = 1.0f / (KNB + 1.0f);
#pragma unroll
  for (int v = 0; v < 8; ++v) {
    int Mo = mtile * 16 + v + 8 * lh;
    int No = base + ntile * 16 + l15;
    size_t o = ((size_t)b * DIM + Mo) * n + No;
    pout[o] = acc[v] * inv + pin[o];
  }
}

// ---------------------------------------------------------------------------
// out[b,0,j] = sum_c w_last[c] * leaky(points[b,c,j])
// ---------------------------------------------------------------------------
__global__ void final_kernel(const float* __restrict__ pin, const float* __restrict__ wl,
                             float* __restrict__ out, int n) {
  int j = blockIdx.x * blockDim.x + threadIdx.x;
  if (j >= BATCH * n) return;
  int b = j / n, i = j - b * n;
  float s = 0.f;
#pragma unroll 4
  for (int c = 0; c < DIM; ++c)
    s += wl[c] * leaky(pin[((size_t)b * DIM + c) * n + i]);
  out[j] = s;
}

// ---------------------------------------------------------------------------
extern "C" void kernel_launch(void* const* d_in, const int* in_sizes, int n_in,
                              void* d_out, int out_size, void* d_ws, size_t ws_size,
                              hipStream_t stream) {
  (void)in_sizes; (void)n_in; (void)out_size; (void)ws_size;
  const float* xyz_in = (const float*)d_in[0];
  const float* w0     = (const float*)d_in[1];
  const float* fn_wc  = (const float*)d_in[2];
  const float* fn_wg  = (const float*)d_in[3];
  const float* res_wc = (const float*)d_in[4];
  const float* res_wg = (const float*)d_in[5];
  const float* w_last = (const float*)d_in[6];
  float* out = (float*)d_out;

  // workspace layout (~18 MB)
  char* ws = (char*)d_ws;
  size_t off = 0;
  auto take = [&](size_t bytes) { char* p = ws + off; off += (bytes + 255) & ~(size_t)255; return p; };
  float* pA   = (float*)take((size_t)BATCH * DIM * N0 * 4);
  float* pB   = (float*)take((size_t)BATCH * DIM * N0 * 4);
  float* xyzA = (float*)take((size_t)BATCH * 3 * (N0 / 4) * 4);
  float* xyzB = (float*)take((size_t)BATCH * 3 * (N0 / 16) * 4);
  int*   idxb = (int*)take((size_t)BATCH * N0 * KNB * 4);
  int*   sidx = (int*)take((size_t)BATCH * (N0 / 4) * 4);

  int n = N0;
  const float* xyz = xyz_in;
  float* pcur = pA;
  float* pnxt = pB;

  // FeatureNet: lift + kNN + 2 res blocks
  lift_kernel<<<dim3((n + 255) / 256, BATCH), 256, 0, stream>>>(xyz, w0, pcur, n);
  knn_kernel<<<dim3((n + 255) / 256, BATCH), 256, 0, stream>>>(xyz, xyz, idxb, n, n);
  for (int j = 0; j < 2; ++j) {
    resblock_kernel<<<dim3(n / NT, BATCH), 256, 0, stream>>>(
        pcur, idxb, fn_wc + (size_t)j * DIM * DIM, fn_wg + (size_t)j * DIM * DIM, pnxt, n);
    float* t = pcur; pcur = pnxt; pnxt = t;
  }

  // 3 x (pool to n/4, then 4 res blocks on the new kNN graph)
  for (int l = 0; l < 3; ++l) {
    int npoint = n / 4;
    float* xnew = (l & 1) ? xyzB : xyzA;
    fps_kernel<<<dim3(BATCH), 256, 0, stream>>>(xyz, sidx, n, npoint);
    gather_xyz_kernel<<<dim3((npoint + 255) / 256, BATCH), 256, 0, stream>>>(xyz, sidx, xnew, n, npoint);
    knn_kernel<<<dim3((npoint + 255) / 256, BATCH), 256, 0, stream>>>(xnew, xyz, idxb, npoint, n);
    pool_max_kernel<<<dim3((npoint + 255) / 256, BATCH), 256, 0, stream>>>(pcur, idxb, pnxt, n, npoint);
    { float* t = pcur; pcur = pnxt; pnxt = t; }
    n = npoint; xyz = xnew;
    knn_kernel<<<dim3((n + 255) / 256, BATCH), 256, 0, stream>>>(xyz, xyz, idxb, n, n);
    for (int i = 0; i < 4; ++i) {
      const float* wcl = res_wc + ((size_t)(l * 4 + i)) * DIM * DIM;
      const float* wgl = res_wg + ((size_t)(l * 4 + i)) * DIM * DIM;
      resblock_kernel<<<dim3(n / NT, BATCH), 256, 0, stream>>>(pcur, idxb, wcl, wgl, pnxt, n);
      float* t = pcur; pcur = pnxt; pnxt = t;
    }
  }

  // final 64 -> 1 projection, n == 64 here; out is [8,1,64] = 512 floats
  final_kernel<<<dim3((BATCH * n + 255) / 256), 256, 0, stream>>>(pcur, w_last, out, n);
}
